// ScaledDotProductAttn_52020643889761
// MI455X (gfx1250) — matrix operand order repaired and checked by reading, not compiled, not run on hardware
//
#include <hip/hip_runtime.h>
#include <hip/hip_bf16.h>

typedef __attribute__((ext_vector_type(16))) _Float16 v16h;
typedef __attribute__((ext_vector_type(8)))  float    v8f;

#define ATT_B 4
#define ATT_H 16
#define ATT_S 2048
#define ATT_D 128
#define BR 64               // query rows per block (4 waves x 16)
#define BC 64               // key rows per tile
#define NWAVES 4
#define NEG_BIG (-3.0e38f)
#define QK_SCALE 0.08838834764831845f   // 1/sqrt(128)

__device__ __forceinline__ v8f wmma_f16(v16h a, v16h b, v8f c) {
    return __builtin_amdgcn_wmma_f32_16x16x32_f16(
        /*neg_a=*/false, a, /*neg_b=*/false, b,
        /*c_mod=*/(short)0, c, /*reuse_a=*/false, /*reuse_b=*/false);
}

__global__ __launch_bounds__(128) void
ScaledDotProductAttn_52020643889761_kernel(const float* __restrict__ Q,
                                           const float* __restrict__ K,
                                           const float* __restrict__ V,
                                           float* __restrict__ O)
{
    __shared__ _Float16 kt[BC][ATT_D];          // K tile, [k][d]
    __shared__ _Float16 vt[ATT_D][BC];          // V tile transposed, [d][k]
    __shared__ _Float16 pt[NWAVES][16][BC];     // wave-private P re-layout buffer

    const int bh    = blockIdx.y;               // fused batch*head
    const int qTile = blockIdx.x;
    const int tid   = threadIdx.x;
    const int wave  = tid >> 5;
    const int lane  = tid & 31;
    const int lh    = lane & 15;                // lane % 16
    const int hi    = lane >> 4;                // 0 or 1 (lane group)

    const size_t base = (size_t)bh * ATT_S * ATT_D;
    const int qRow0 = qTile * BR;

    // ---- load Q as A-layout f16 fragments, pre-scaled by 1/sqrt(D) ----
    // A 16x32 f16 fragment, this kernel's contiguous k order: element h -> K = kc*32 + hi*16 + h
    v16h qf[4];
    {
        const float* qp = Q + base + (size_t)(qRow0 + wave * 16 + lh) * ATT_D;
        #pragma unroll
        for (int kc = 0; kc < 4; ++kc) {
            const int d0 = kc * 32 + hi * 16;
            #pragma unroll
            for (int h = 0; h < 8; ++h) {
                qf[kc][h]     = (_Float16)(qp[d0 + h]      * QK_SCALE);
                qf[kc][h + 8] = (_Float16)(qp[d0 + 8 + h] * QK_SCALE);
            }
        }
    }

    // ---- accumulators: O in C-layout (8 n-tiles of 16 cols), row stats ----
    v8f o[8];
    #pragma unroll
    for (int n = 0; n < 8; ++n) o[n] = (v8f){0.f,0.f,0.f,0.f,0.f,0.f,0.f,0.f};
    float mrow[8], lrow[8];
    #pragma unroll
    for (int g = 0; g < 8; ++g) { mrow[g] = NEG_BIG; lrow[g] = 0.f; }

    const int qrBase = qRow0 + wave * 16 + hi * 8;   // C-layout row for g=0
    const int jTiles = qTile + 1;                    // causal: stop at diagonal

    for (int j = 0; j < jTiles; ++j) {
        const int kRow0 = j * BC;

        __syncthreads();   // previous-iteration LDS readers done
        // cooperative K/V tile load, f32 -> f16, V transposed
        for (int e = tid * 4; e < BC * ATT_D; e += 128 * 4) {
            const int kr = e >> 7;           // / ATT_D
            const int d  = e & (ATT_D - 1);
            const float4 kv = *(const float4*)(K + base + (size_t)(kRow0 + kr) * ATT_D + d);
            kt[kr][d + 0] = (_Float16)kv.x;
            kt[kr][d + 1] = (_Float16)kv.y;
            kt[kr][d + 2] = (_Float16)kv.z;
            kt[kr][d + 3] = (_Float16)kv.w;
            const float4 vv = *(const float4*)(V + base + (size_t)(kRow0 + kr) * ATT_D + d);
            vt[d + 0][kr] = (_Float16)vv.x;
            vt[d + 1][kr] = (_Float16)vv.y;
            vt[d + 2][kr] = (_Float16)vv.z;
            vt[d + 3][kr] = (_Float16)vv.w;
        }
        // prefetch next K/V tile while we compute on this one
        if (j + 1 < jTiles) {
            __builtin_prefetch(K + base + (size_t)(kRow0 + BC) * ATT_D + tid * 64, 0, 0);
            __builtin_prefetch(V + base + (size_t)(kRow0 + BC) * ATT_D + tid * 64, 0, 0);
        }
        __syncthreads();

        // ---- S = (Q * scale) . K^T : 16x64 per wave, 16 WMMAs ----
        v8f s[4];
        #pragma unroll
        for (int t = 0; t < 4; ++t) {
            v8f acc = (v8f){0.f,0.f,0.f,0.f,0.f,0.f,0.f,0.f};
            #pragma unroll
            for (int kc = 0; kc < 4; ++kc) {
                // B 32x16 fragment, same contiguous k order as qf: col N = t*16+lh (key row), half h -> d = kc*32 + hi*16 + h
                v16h kf = *(const v16h*)&kt[t * 16 + lh][kc * 32 + hi * 16];
                acc = wmma_f16(qf[kc], kf, acc);
            }
            s[t] = acc;
        }

        // ---- causal mask ----
        #pragma unroll
        for (int t = 0; t < 4; ++t) {
            const int kcol = kRow0 + t * 16 + lh;
            #pragma unroll
            for (int g = 0; g < 8; ++g)
                if (kcol > qrBase + g) s[t][g] = NEG_BIG;
        }

        // ---- online softmax: row max over 64 cols (16-lane butterfly) ----
        float mnew[8];
        #pragma unroll
        for (int g = 0; g < 8; ++g) {
            float v = fmaxf(fmaxf(s[0][g], s[1][g]), fmaxf(s[2][g], s[3][g]));
            #pragma unroll
            for (int off = 8; off >= 1; off >>= 1)
                v = fmaxf(v, __shfl_xor(v, off, 32));
            mnew[g] = fmaxf(mrow[g], v);
        }

        float rsum[8];
        #pragma unroll
        for (int g = 0; g < 8; ++g) rsum[g] = 0.f;
        #pragma unroll
        for (int t = 0; t < 4; ++t) {
            #pragma unroll
            for (int g = 0; g < 8; ++g) {
                const float p = __expf(s[t][g] - mnew[g]);
                s[t][g] = p;
                rsum[g] += p;
            }
        }
        #pragma unroll
        for (int g = 0; g < 8; ++g) {
            float v = rsum[g];
            #pragma unroll
            for (int off = 8; off >= 1; off >>= 1)
                v += __shfl_xor(v, off, 32);
            const float corr = __expf(mrow[g] - mnew[g]);
            lrow[g] = lrow[g] * corr + v;
            mrow[g] = mnew[g];
            #pragma unroll
            for (int n = 0; n < 8; ++n) o[n][g] *= corr;   // rescale O row g
        }

        // ---- re-layout P: C layout -> wave-private LDS -> A fragments ----
        #pragma unroll
        for (int t = 0; t < 4; ++t)
            #pragma unroll
            for (int g = 0; g < 8; ++g)
                pt[wave][g + hi * 8][t * 16 + lh] = (_Float16)s[t][g];

        asm volatile("s_wait_dscnt 0x0" ::: "memory");  // same-wave DS RAW fence

        v16h pf[2];
        #pragma unroll
        for (int kc = 0; kc < 2; ++kc) {
            const int k0 = kc * 32 + hi * 16;
            #pragma unroll
            for (int h = 0; h < 8; ++h) {
                pf[kc][h]     = pt[wave][lh][k0 + h];
                pf[kc][h + 8] = pt[wave][lh][k0 + 8 + h];
            }
        }

        // ---- O += P(16x64) . V(64x128) : 16 WMMAs ----
        #pragma unroll
        for (int n = 0; n < 8; ++n) {
            #pragma unroll
            for (int kc = 0; kc < 2; ++kc) {
                // B: col N = d = n*16+lh, half h -> k = kc*32 + hi*16 + h
                v16h vf = *(const v16h*)&vt[n * 16 + lh][kc * 32 + hi * 16];
                o[n] = wmma_f16(pf[kc], vf, o[n]);
            }
        }
    }

    // ---- normalize & store (C layout -> [S,D]) ----
    float* op = O + base;
    #pragma unroll
    for (int g = 0; g < 8; ++g) {
        const float inv_l = 1.0f / lrow[g];
        const size_t row = (size_t)(qrBase + g) * ATT_D;
        #pragma unroll
        for (int n = 0; n < 8; ++n)
            op[row + n * 16 + lh] = o[n][g] * inv_l;
    }
}

extern "C" void kernel_launch(void* const* d_in, const int* in_sizes, int n_in,
                              void* d_out, int out_size, void* d_ws, size_t ws_size,
                              hipStream_t stream) {
    const float* q = (const float*)d_in[0];
    const float* k = (const float*)d_in[1];
    const float* v = (const float*)d_in[2];
    // d_in[3] is the causal mask; causality is computed analytically in-kernel.
    float* out = (float*)d_out;
    dim3 grid(ATT_S / BR, ATT_B * ATT_H);
    dim3 block(128);
    hipLaunchKernelGGL(ScaledDotProductAttn_52020643889761_kernel,
                       grid, block, 0, stream, q, k, v, out);
}
